// TernaryLinear_33818572489121
// MI455X (gfx1250) — matrix operand order repaired and checked by reading, not compiled, not run on hardware
//
#include <hip/hip_runtime.h>

// ---------------------------------------------------------------------------
// TernaryLinear for MI455X (gfx1250):  out = x @ ternary(w).T + round(bias)
// Two-phase: (1) split x -> bf16 hi/lo, quantize w -> ternary bf16 (in d_ws)
//            (2) double-buffered WMMA GEMM with async global->LDS staging
// Falls back to a fused single-pass kernel if the workspace is too small.
// ---------------------------------------------------------------------------

typedef __attribute__((ext_vector_type(16))) __bf16 v16bf;
typedef __attribute__((ext_vector_type(8)))  __bf16 v8bf;
typedef __attribute__((ext_vector_type(4)))  __bf16 v4bf;
typedef __attribute__((ext_vector_type(8)))  float  v8f;
typedef __attribute__((ext_vector_type(4)))  float  v4f;
typedef __attribute__((ext_vector_type(4)))  int    v4i;

#define MDIM 8192
#define NDIM 4096
#define KDIM 4096
#define BM   128
#define BN   128
#define BK   64
#define LDSP 72                 // padded LDS row stride (bf16 elems) -> conflict-free b128 reads
#define PANEL (BM * LDSP)       // 9216 elements per staged matrix panel

#if defined(__AMDGCN__) && defined(__has_builtin)
# if __has_builtin(__builtin_amdgcn_global_load_async_to_lds_b128)
#  define HAVE_ASYNC_COPY 1
# endif
#endif

union Frag { v16bf v; v8bf h[2]; };

// 16-byte global->LDS copy: async (ASYNCcnt) when available, sync otherwise.
__device__ __forceinline__ void copy16(const __bf16* g, __bf16* l) {
#if defined(HAVE_ASYNC_COPY)
    __builtin_amdgcn_global_load_async_to_lds_b128(
        (__attribute__((address_space(1))) v4i*)g,
        (__attribute__((address_space(3))) v4i*)l,
        0, 0);
#else
    *(v8bf*)l = *(const v8bf*)g;
#endif
}

__device__ __forceinline__ void wait_async() {
#if defined(HAVE_ASYNC_COPY)
    asm volatile("s_wait_asynccnt 0x0" ::: "memory");
#endif
}

// ---------------- phase 1: precision split / quantization -------------------

__global__ __launch_bounds__(256)
void prep_x(const float* __restrict__ X, __bf16* __restrict__ Xhi,
            __bf16* __restrict__ Xlo) {
    size_t idx = ((size_t)blockIdx.x * 256u + threadIdx.x) * 4u;
    v4f xv = *(const v4f*)(X + idx);
    v4bf h, l;
#pragma unroll
    for (int j = 0; j < 4; ++j) {
        float  f  = xv[j];
        __bf16 hb = (__bf16)f;
        h[j] = hb;
        l[j] = (__bf16)(f - (float)hb);
    }
    *(v4bf*)(Xhi + idx) = h;
    *(v4bf*)(Xlo + idx) = l;
}

__global__ __launch_bounds__(256)
void prep_w(const float* __restrict__ W, __bf16* __restrict__ Wq) {
    size_t idx = ((size_t)blockIdx.x * 256u + threadIdx.x) * 4u;
    v4f wv = *(const v4f*)(W + idx);
    v4bf q;
#pragma unroll
    for (int j = 0; j < 4; ++j) {
        float r = rintf(wv[j]);     // half-to-even == jnp.round
        q[j] = (__bf16)((r > 0.f) ? 1.f : ((r < 0.f) ? -1.f : 0.f));
    }
    *(v4bf*)(Wq + idx) = q;
}

// ---------------- phase 2: double-buffered WMMA GEMM ------------------------

__global__ __launch_bounds__(256)
void gemm_staged(const __bf16* __restrict__ Xhi, const __bf16* __restrict__ Xlo,
                 const __bf16* __restrict__ Wq, const float* __restrict__ Bv,
                 float* __restrict__ Out)
{
    extern __shared__ __bf16 smem[];   // 2 bufs x {Ahi, Alo, B} x PANEL = 110592 B

    const unsigned tid  = threadIdx.x;
    const unsigned lane = tid & 31u;
    const unsigned wid  = tid >> 5;
    const unsigned l16  = lane & 15u;
    const unsigned half = lane >> 4;

    const unsigned mbase = (wid & 1u) * 64u;   // 2 waves across M
    const unsigned nbase = (wid >> 1) * 32u;   // 4 waves across N

    const unsigned tile_m = blockIdx.y * BM;
    const unsigned tile_n = blockIdx.x * BN;

    // stage one K-panel (Ahi, Alo, B) into buffer `buf` via async copies
    auto stage = [&](unsigned buf, unsigned k0) {
        __bf16* dst = smem + buf * (3u * PANEL);
#pragma unroll
        for (int i = 0; i < 4; ++i) {
            unsigned c   = tid + (unsigned)i * 256u;   // 1024 16B chunks per panel
            unsigned row = c >> 3;
            unsigned ch  = (c & 7u) * 8u;
            size_t   ga  = (size_t)(tile_m + row) * KDIM + k0 + ch;
            size_t   gb  = (size_t)(tile_n + row) * KDIM + k0 + ch;
            unsigned lo  = row * LDSP + ch;
            copy16(Xhi + ga, dst + lo);
            copy16(Xlo + ga, dst + PANEL + lo);
            copy16(Wq + gb,  dst + 2u * PANEL + lo);
        }
    };

    v8f acc[4][2];
#pragma unroll
    for (int mm = 0; mm < 4; ++mm)
#pragma unroll
        for (int nn = 0; nn < 2; ++nn)
            acc[mm][nn] = (v8f){0.f, 0.f, 0.f, 0.f, 0.f, 0.f, 0.f, 0.f};

    stage(0u, 0u);

    const unsigned NK = KDIM / BK;
    for (unsigned k = 0; k < NK; ++k) {
        const unsigned buf = k & 1u;
        wait_async();           // my async copies into buf are done
        __syncthreads();        // everyone's copies into buf are done;
                                // everyone finished reading buf^1
        if (k + 1u < NK) stage(buf ^ 1u, (k + 1u) * BK);

        const __bf16* Ahi = smem + buf * (3u * PANEL);
        const __bf16* Alo = Ahi + PANEL;
        const __bf16* Bb  = Ahi + 2u * PANEL;

#pragma unroll
        for (int sdx = 0; sdx < 2; ++sdx) {
            const unsigned ks = (unsigned)sdx * 32u;

            Frag bf[2];
#pragma unroll
            for (int nn = 0; nn < 2; ++nn) {
                const __bf16* bp = &Bb[(nbase + nn * 16u + l16) * LDSP + ks + half * 8u];
                bf[nn].h[0] = *(const v8bf*)(bp);
                bf[nn].h[1] = *(const v8bf*)(bp + 16);
            }
#pragma unroll
            for (int mm = 0; mm < 4; ++mm) {
                const __bf16* ap = &Ahi[(mbase + mm * 16u + l16) * LDSP + ks + half * 8u];
                Frag a;
                a.h[0] = *(const v8bf*)(ap);
                a.h[1] = *(const v8bf*)(ap + 16);
#pragma unroll
                for (int nn = 0; nn < 2; ++nn)
                    acc[mm][nn] = __builtin_amdgcn_wmma_f32_16x16x32_bf16(
                        false, a.v, false, bf[nn].v, (short)0, acc[mm][nn], false, false);
            }
#pragma unroll
            for (int mm = 0; mm < 4; ++mm) {
                const __bf16* ap = &Alo[(mbase + mm * 16u + l16) * LDSP + ks + half * 8u];
                Frag a;
                a.h[0] = *(const v8bf*)(ap);
                a.h[1] = *(const v8bf*)(ap + 16);
#pragma unroll
                for (int nn = 0; nn < 2; ++nn)
                    acc[mm][nn] = __builtin_amdgcn_wmma_f32_16x16x32_bf16(
                        false, a.v, false, bf[nn].v, (short)0, acc[mm][nn], false, false);
            }
        }
        __syncthreads();        // all reads of buf done before it is restaged
    }

    // epilogue: quantized bias + store (C/D: col N = l16, VGPR r -> M = r + 8*half)
#pragma unroll
    for (int nn = 0; nn < 2; ++nn) {
        const unsigned o = tile_n + nbase + nn * 16u + l16;
        float bq = rintf(Bv[o]);
        bq = fminf(8.0f, fmaxf(-8.0f, bq));
#pragma unroll
        for (int mm = 0; mm < 4; ++mm) {
            const unsigned row0 = tile_m + mbase + mm * 16u + half * 8u;
#pragma unroll
            for (int r = 0; r < 8; ++r)
                Out[(size_t)(row0 + r) * NDIM + o] = acc[mm][nn][r] + bq;
        }
    }
}

// ---------------- fallback: fused single-pass kernel ------------------------

__global__ __launch_bounds__(256)
void ternary_linear_fused(const float* __restrict__ X, const float* __restrict__ W,
                          const float* __restrict__ Bv, float* __restrict__ Out)
{
    __shared__ __bf16 sAhi[BM * LDSP];
    __shared__ __bf16 sAlo[BM * LDSP];
    __shared__ __bf16 sB  [BN * LDSP];

    const unsigned tid  = threadIdx.x;
    const unsigned lane = tid & 31u;
    const unsigned wid  = tid >> 5;
    const unsigned l16  = lane & 15u;
    const unsigned half = lane >> 4;
    const unsigned mbase = (wid & 1u) * 64u;
    const unsigned nbase = (wid >> 1) * 32u;
    const unsigned tile_m = blockIdx.y * BM;
    const unsigned tile_n = blockIdx.x * BN;

    v8f acc[4][2];
#pragma unroll
    for (int mm = 0; mm < 4; ++mm)
#pragma unroll
        for (int nn = 0; nn < 2; ++nn)
            acc[mm][nn] = (v8f){0.f, 0.f, 0.f, 0.f, 0.f, 0.f, 0.f, 0.f};

    for (unsigned k0 = 0; k0 < KDIM; k0 += BK) {
        __syncthreads();
#pragma unroll
        for (int i = 0; i < 8; ++i) {
            unsigned s   = tid + i * 256u;
            unsigned row = s >> 4;
            unsigned c4  = (s & 15u) << 2;
            v4f xv = *(const v4f*)(X + (size_t)(tile_m + row) * KDIM + k0 + c4);
            v4bf h, l;
#pragma unroll
            for (int j = 0; j < 4; ++j) {
                float  f  = xv[j];
                __bf16 hb = (__bf16)f;
                h[j] = hb;
                l[j] = (__bf16)(f - (float)hb);
            }
            *(v4bf*)(&sAhi[row * LDSP + c4]) = h;
            *(v4bf*)(&sAlo[row * LDSP + c4]) = l;
        }
#pragma unroll
        for (int i = 0; i < 8; ++i) {
            unsigned s   = tid + i * 256u;
            unsigned row = s >> 4;
            unsigned c4  = (s & 15u) << 2;
            v4f wv = *(const v4f*)(W + (size_t)(tile_n + row) * KDIM + k0 + c4);
            v4bf q;
#pragma unroll
            for (int j = 0; j < 4; ++j) {
                float r = rintf(wv[j]);
                q[j] = (__bf16)((r > 0.f) ? 1.f : ((r < 0.f) ? -1.f : 0.f));
            }
            *(v4bf*)(&sB[row * LDSP + c4]) = q;
        }
        __syncthreads();
#pragma unroll
        for (int sdx = 0; sdx < 2; ++sdx) {
            const unsigned ks = (unsigned)sdx * 32u;
            Frag bf[2];
#pragma unroll
            for (int nn = 0; nn < 2; ++nn) {
                const __bf16* bp = &sB[(nbase + nn * 16u + l16) * LDSP + ks + half * 8u];
                bf[nn].h[0] = *(const v8bf*)(bp);
                bf[nn].h[1] = *(const v8bf*)(bp + 16);
            }
#pragma unroll
            for (int mm = 0; mm < 4; ++mm) {
                const __bf16* ap = &sAhi[(mbase + mm * 16u + l16) * LDSP + ks + half * 8u];
                Frag a;
                a.h[0] = *(const v8bf*)(ap);
                a.h[1] = *(const v8bf*)(ap + 16);
#pragma unroll
                for (int nn = 0; nn < 2; ++nn)
                    acc[mm][nn] = __builtin_amdgcn_wmma_f32_16x16x32_bf16(
                        false, a.v, false, bf[nn].v, (short)0, acc[mm][nn], false, false);
            }
#pragma unroll
            for (int mm = 0; mm < 4; ++mm) {
                const __bf16* ap = &sAlo[(mbase + mm * 16u + l16) * LDSP + ks + half * 8u];
                Frag a;
                a.h[0] = *(const v8bf*)(ap);
                a.h[1] = *(const v8bf*)(ap + 16);
#pragma unroll
                for (int nn = 0; nn < 2; ++nn)
                    acc[mm][nn] = __builtin_amdgcn_wmma_f32_16x16x32_bf16(
                        false, a.v, false, bf[nn].v, (short)0, acc[mm][nn], false, false);
            }
        }
    }
#pragma unroll
    for (int nn = 0; nn < 2; ++nn) {
        const unsigned o = tile_n + nbase + nn * 16u + l16;
        float bq = rintf(Bv[o]);
        bq = fminf(8.0f, fmaxf(-8.0f, bq));
#pragma unroll
        for (int mm = 0; mm < 4; ++mm) {
            const unsigned row0 = tile_m + mbase + mm * 16u + half * 8u;
#pragma unroll
            for (int r = 0; r < 8; ++r)
                Out[(size_t)(row0 + r) * NDIM + o] = acc[mm][nn][r] + bq;
        }
    }
}

// ---------------------------------------------------------------------------

extern "C" void kernel_launch(void* const* d_in, const int* in_sizes, int n_in,
                              void* d_out, int out_size, void* d_ws, size_t ws_size,
                              hipStream_t stream) {
    const float* X  = (const float*)d_in[0];   // [8192, 4096]
    const float* W  = (const float*)d_in[1];   // [4096, 4096]
    const float* Bv = (const float*)d_in[2];   // [4096]
    float*       Y  = (float*)d_out;           // [8192, 4096]

    const size_t xElems = (size_t)MDIM * KDIM;         // 33.5M
    const size_t wElems = (size_t)NDIM * KDIM;         // 16.8M
    const size_t wsNeeded = (2 * xElems + wElems) * sizeof(__bf16);  // 160 MB

    dim3 grid(NDIM / BN, MDIM / BM);   // (32, 64)
    dim3 block(256);

    if (ws_size >= wsNeeded) {
        __bf16* Xhi = (__bf16*)d_ws;
        __bf16* Xlo = Xhi + xElems;
        __bf16* Wq  = Xlo + xElems;
        hipLaunchKernelGGL(prep_x, dim3((unsigned)(xElems / 4 / 256)), block, 0, stream,
                           X, Xhi, Xlo);
        hipLaunchKernelGGL(prep_w, dim3((unsigned)(wElems / 4 / 256)), block, 0, stream,
                           W, Wq);
        const unsigned smemBytes = 2u * 3u * PANEL * (unsigned)sizeof(__bf16); // 110592
        hipLaunchKernelGGL(gemm_staged, grid, block, smemBytes, stream,
                           Xhi, Xlo, Wq, Bv, Y);
    } else {
        hipLaunchKernelGGL(ternary_linear_fused, grid, block, 0, stream, X, W, Bv, Y);
    }

    (void)in_sizes; (void)n_in; (void)out_size;
}
